// MultiHeadSelfAttention_32212254720501
// MI455X (gfx1250) — matrix operand order repaired and checked by reading, not compiled, not run on hardware
//
#include <hip/hip_runtime.h>
#include <hip/hip_bf16.h>

// MHSA forward for MI455X (gfx1250): bf16 WMMA, f32 accumulate.
// B=2, T=2048, D=1024, H=16, dk=64.

#define D_MODEL 1024
#define NUM_HEADS 16
#define D_KEY 64
#define SEQ_T 2048
#define BATCH 2
#define ROWS (BATCH * SEQ_T) // 4096

typedef __bf16 bf16_t;
typedef __attribute__((ext_vector_type(16))) __bf16 v16bf;
typedef __attribute__((ext_vector_type(8)))  __bf16 v8bf;
typedef __attribute__((ext_vector_type(8)))  float  v8f;

static __device__ __forceinline__ unsigned short f2bf_u16(float f) {
  union { float f; unsigned u; } v; v.f = f;
  unsigned r = v.u + 0x7FFFu + ((v.u >> 16) & 1u); // round-to-nearest-even
  return (unsigned short)(r >> 16);
}
static __device__ __forceinline__ bf16_t f2bf(float f) {
  unsigned short s = f2bf_u16(f);
  return __builtin_bit_cast(bf16_t, s);
}

// ---- WMMA fragment loaders (wave32, 16-bit data; this code's contiguous k order, shared by A and B) ----
// A-matrix 16x32 (MxK): lane L holds row M=L&15.
//   lanes 0-15:  halfs[0..15]=K0..15
//   lanes 16-31: halfs[0..15]=K16..31
static __device__ __forceinline__ v16bf load_frag_a(const bf16_t* tile, int ld, int lane) {
  const bf16_t* row = tile + (size_t)(lane & 15) * ld;
  const int sel = (lane >> 4) << 4; // 0 or 16
  v8bf lo = *(const v8bf*)(row + sel);
  v8bf hi = *(const v8bf*)(row + sel + 8);
  return __builtin_shufflevector(lo, hi, 0,1,2,3,4,5,6,7,8,9,10,11,12,13,14,15);
}
// B-matrix 32x16 (KxN) where Bmat[k,n] = src[n*ld + k] (i.e. src holds B^T rows
// contiguously — exactly what y = x*W^T and S^T = K*Q^T need).
//   lane L: N=L&15; lanes 0-15 hold K=0..15, lanes 16-31 hold K=16..31.
static __device__ __forceinline__ v16bf load_frag_b(const bf16_t* tile, int ld, int lane) {
  return *(const v16bf*)(tile + (size_t)(lane & 15) * ld + ((lane >> 4) << 4));
}

static __device__ __forceinline__ v8f wmma_bf16(v16bf a, v16bf b, v8f c) {
  return __builtin_amdgcn_wmma_f32_16x16x32_bf16(false, a, false, b, (short)0, c, false, false);
}

// ---------------------------------------------------------------------------
// Kernel 1: f32 -> bf16 conversion (grid covers exactly n/4 quads)
// ---------------------------------------------------------------------------
__global__ void __launch_bounds__(256) cvt_f32_bf16(const float* __restrict__ in,
                                                    bf16_t* __restrict__ out, int n) {
  int i = (blockIdx.x * 256 + threadIdx.x) * 4;
  if (i + 3 < n) {
    float4 v = *(const float4*)(in + i);
    union { unsigned short s[4]; uint2 u; } o;
    o.s[0] = f2bf_u16(v.x); o.s[1] = f2bf_u16(v.y);
    o.s[2] = f2bf_u16(v.z); o.s[3] = f2bf_u16(v.w);
    *(uint2*)(out + i) = o.u;
  }
}

// ---------------------------------------------------------------------------
// Kernel 2: QKV projections. One wave computes a 32(M) x 64(N) tile.
// Block 256 threads = 8 waves arranged 2(M) x 4(N) -> block tile 64 x 256.
// Grid: (N/256, M/64, 3) = (4, 64, 3).
// z=0: Q (scaled 1/sqrt(dk)) -> (B,H,T,64); z=1: K -> (B,H,T,64);
// z=2: V -> transposed (B,H,64,T) so the PV GEMM A-frags are row loads.
// ---------------------------------------------------------------------------
__global__ void __launch_bounds__(256) qkv_gemm(const bf16_t* __restrict__ X,
    const bf16_t* __restrict__ Wq, const bf16_t* __restrict__ Wk,
    const bf16_t* __restrict__ Wv,
    bf16_t* __restrict__ Qb, bf16_t* __restrict__ Kb, bf16_t* __restrict__ Vtb) {
  const int lane = threadIdx.x & 31;
  const int wid  = threadIdx.x >> 5;
  const int m0 = blockIdx.y * 64 + (wid >> 2) * 32;
  const int n0 = blockIdx.x * 256 + (wid & 3) * 64;
  const int z  = blockIdx.z;
  const bf16_t* W = (z == 0) ? Wq : (z == 1) ? Wk : Wv;

  v8f acc[2][4];
  #pragma unroll
  for (int mi = 0; mi < 2; ++mi)
    #pragma unroll
    for (int j = 0; j < 4; ++j)
      #pragma unroll
      for (int vi = 0; vi < 8; ++vi) acc[mi][j][vi] = 0.0f;

  for (int k0 = 0; k0 < D_MODEL; k0 += 32) {
    v16bf a0 = load_frag_a(X + (size_t)m0 * D_MODEL + k0, D_MODEL, lane);
    v16bf a1 = load_frag_a(X + (size_t)(m0 + 16) * D_MODEL + k0, D_MODEL, lane);
    #pragma unroll
    for (int j = 0; j < 4; ++j) {
      v16bf b = load_frag_b(W + (size_t)(n0 + j * 16) * D_MODEL + k0, D_MODEL, lane);
      acc[0][j] = wmma_bf16(a0, b, acc[0][j]);
      acc[1][j] = wmma_bf16(a1, b, acc[1][j]);
    }
  }

  const float scale = (z == 0) ? 0.125f : 1.0f; // 1/sqrt(64) folded into Q
  const int mg = (lane >> 4) << 3;
  #pragma unroll
  for (int mi = 0; mi < 2; ++mi) {
    #pragma unroll
    for (int j = 0; j < 4; ++j) {
      const int n = n0 + j * 16 + (lane & 15);
      const int h = n >> 6, dd = n & 63;
      #pragma unroll
      for (int vi = 0; vi < 8; ++vi) {
        const int m = m0 + mi * 16 + vi + mg;       // C-frag: M = vi (+8 hi lanes)
        const int bb = m >> 11, t = m & 2047;
        const bf16_t hv = f2bf(acc[mi][j][vi] * scale);
        if (z == 2) {
          Vtb[((size_t)(bb * NUM_HEADS + h) * D_KEY + dd) * SEQ_T + t] = hv;
        } else {
          bf16_t* dst = (z == 0) ? Qb : Kb;
          dst[((size_t)(bb * NUM_HEADS + h) * SEQ_T + t) * D_KEY + dd] = hv;
        }
      }
    }
  }
}

// ---------------------------------------------------------------------------
// Kernel 3: causal flash attention. One wave per 16-query tile.
// S^T = K_tile(16x64) * Q^T(64x16): lane owns query column -> softmax
// reductions are in-lane + one shfl_xor(16). P staged in wave-private LDS
// (stride 34 halfs, conflict-free) to form the B-frag for O^T = V^T * P^T.
// Block 128 threads = 4 waves; grid (T/16/4, H, B) = (32, 16, 2).
// ---------------------------------------------------------------------------
__global__ void __launch_bounds__(128) attn_kernel(const bf16_t* __restrict__ Qb,
    const bf16_t* __restrict__ Kb, const bf16_t* __restrict__ Vtb,
    bf16_t* __restrict__ Ab) {
  const int lane = threadIdx.x & 31;
  const int wid  = threadIdx.x >> 5;
  const int q0   = (blockIdx.x * 4 + wid) * 16;
  const int h = blockIdx.y, b = blockIdx.z;

  __shared__ unsigned short plds[4][16 * 34];
  unsigned short* pw = &plds[wid][0];

  const bf16_t* qbase = Qb + ((size_t)(b * NUM_HEADS + h) * SEQ_T + q0) * D_KEY;
  const bf16_t* kbase = Kb + (size_t)(b * NUM_HEADS + h) * SEQ_T * D_KEY;
  const bf16_t* vbase = Vtb + (size_t)(b * NUM_HEADS + h) * D_KEY * SEQ_T;

  // Q^T B-fragments, resident for the whole loop (dk=64 -> two K=32 steps).
  const v16bf bq0 = load_frag_b(qbase, D_KEY, lane);
  const v16bf bq1 = load_frag_b(qbase + 32, D_KEY, lane);

  float mrun = -3.0e38f, lrun = 0.0f;
  v8f oacc[4];
  #pragma unroll
  for (int dt = 0; dt < 4; ++dt)
    #pragma unroll
    for (int vi = 0; vi < 8; ++vi) oacc[dt][vi] = 0.0f;

  const int q  = q0 + (lane & 15);   // this lane's query index
  const int mg = (lane >> 4) << 3;   // C-frag row group offset (0 or 8)

  for (int j0 = 0; j0 < q0 + 16; j0 += 32) {
    float s[2][8];
    #pragma unroll
    for (int sub = 0; sub < 2; ++sub) {
      const int jb = j0 + sub * 16;
      if (jb < q0 + 16) {                       // wave-uniform
        v16bf ka0 = load_frag_a(kbase + (size_t)jb * D_KEY, D_KEY, lane);
        v16bf ka1 = load_frag_a(kbase + (size_t)jb * D_KEY + 32, D_KEY, lane);
        v8f c;
        #pragma unroll
        for (int vi = 0; vi < 8; ++vi) c[vi] = 0.0f;
        c = wmma_bf16(ka0, bq0, c);
        c = wmma_bf16(ka1, bq1, c);
        #pragma unroll
        for (int vi = 0; vi < 8; ++vi) {
          const int key = jb + vi + mg;
          s[sub][vi] = (key > q) ? -3.0e38f : c[vi];   // causal mask
        }
      } else {
        #pragma unroll
        for (int vi = 0; vi < 8; ++vi) s[sub][vi] = -3.0e38f;
      }
    }

    // online softmax: in-lane max over 16 vals + partner-lane combine
    float lmax = s[0][0];
    #pragma unroll
    for (int vi = 1; vi < 8; ++vi) lmax = fmaxf(lmax, s[0][vi]);
    #pragma unroll
    for (int vi = 0; vi < 8; ++vi) lmax = fmaxf(lmax, s[1][vi]);
    lmax = fmaxf(lmax, __shfl_xor(lmax, 16));
    const float mnew = fmaxf(mrun, lmax);
    const float rescale = __expf(mrun - mnew);
    mrun = mnew;
    lrun *= rescale;
    #pragma unroll
    for (int dt = 0; dt < 4; ++dt)
      #pragma unroll
      for (int vi = 0; vi < 8; ++vi) oacc[dt][vi] *= rescale;

    // P = exp(S - m) -> LDS tile [query][key], stride 34 halfs
    #pragma unroll
    for (int sub = 0; sub < 2; ++sub)
      #pragma unroll
      for (int vi = 0; vi < 8; ++vi) {
        const float p = __expf(s[sub][vi] - mnew);
        lrun += p;
        pw[(lane & 15) * 34 + sub * 16 + vi + mg] = f2bf_u16(p);
      }
    asm volatile("s_wait_dscnt 0x0" ::: "memory"); // wave-private LDS: no barrier

    // re-load P as the B-fragment of O^T = V^T * P^T
    union { unsigned u[8]; v16bf v; } cv;
    const unsigned* src =
        (const unsigned*)(pw + (lane & 15) * 34 + ((lane >> 4) << 4));
    #pragma unroll
    for (int i = 0; i < 8; ++i) cv.u[i] = src[i];
    const v16bf bp = cv.v;

    #pragma unroll
    for (int dt = 0; dt < 4; ++dt) {
      v16bf av = load_frag_a(vbase + (size_t)(dt * 16) * SEQ_T + j0, SEQ_T, lane);
      oacc[dt] = wmma_bf16(av, bp, oacc[dt]);
    }
  }

  const float ltot = lrun + __shfl_xor(lrun, 16);
  const float inv = 1.0f / ltot;
  const int t = q0 + (lane & 15);
  bf16_t* arow = Ab + ((size_t)b * SEQ_T + t) * D_MODEL + h * D_KEY;
  #pragma unroll
  for (int dt = 0; dt < 4; ++dt)
    #pragma unroll
    for (int vi = 0; vi < 8; ++vi)
      arow[dt * 16 + vi + mg] = f2bf(oacc[dt][vi] * inv);
}

// ---------------------------------------------------------------------------
// Kernel 4: output projection y = A * Wo^T, f32 result. Same tiling as QKV.
// ---------------------------------------------------------------------------
__global__ void __launch_bounds__(256) proj_gemm(const bf16_t* __restrict__ A,
    const bf16_t* __restrict__ Wo, float* __restrict__ out) {
  const int lane = threadIdx.x & 31;
  const int wid  = threadIdx.x >> 5;
  const int m0 = blockIdx.y * 64 + (wid >> 2) * 32;
  const int n0 = blockIdx.x * 256 + (wid & 3) * 64;

  v8f acc[2][4];
  #pragma unroll
  for (int mi = 0; mi < 2; ++mi)
    #pragma unroll
    for (int j = 0; j < 4; ++j)
      #pragma unroll
      for (int vi = 0; vi < 8; ++vi) acc[mi][j][vi] = 0.0f;

  for (int k0 = 0; k0 < D_MODEL; k0 += 32) {
    v16bf a0 = load_frag_a(A + (size_t)m0 * D_MODEL + k0, D_MODEL, lane);
    v16bf a1 = load_frag_a(A + (size_t)(m0 + 16) * D_MODEL + k0, D_MODEL, lane);
    #pragma unroll
    for (int j = 0; j < 4; ++j) {
      v16bf b = load_frag_b(Wo + (size_t)(n0 + j * 16) * D_MODEL + k0, D_MODEL, lane);
      acc[0][j] = wmma_bf16(a0, b, acc[0][j]);
      acc[1][j] = wmma_bf16(a1, b, acc[1][j]);
    }
  }

  const int mg = (lane >> 4) << 3;
  #pragma unroll
  for (int mi = 0; mi < 2; ++mi)
    #pragma unroll
    for (int j = 0; j < 4; ++j) {
      const int n = n0 + j * 16 + (lane & 15);
      #pragma unroll
      for (int vi = 0; vi < 8; ++vi) {
        const int m = m0 + mi * 16 + vi + mg;
        out[(size_t)m * D_MODEL + n] = acc[mi][j][vi];
      }
    }
}

// ---------------------------------------------------------------------------
extern "C" void kernel_launch(void* const* d_in, const int* in_sizes, int n_in,
                              void* d_out, int out_size, void* d_ws, size_t ws_size,
                              hipStream_t stream) {
  const float* x  = (const float*)d_in[0];
  const float* Wq = (const float*)d_in[1];
  const float* Wk = (const float*)d_in[2];
  const float* Wv = (const float*)d_in[3];
  const float* Wo = (const float*)d_in[4];

  char* ws = (char*)d_ws;
  size_t off = 0;
  auto alloc = [&](size_t bytes) -> void* {
    void* p = ws + off;
    off += (bytes + 255) & ~(size_t)255;
    return p;
  };
  const size_t nX = (size_t)ROWS * D_MODEL;      // 4,194,304
  const size_t nW = (size_t)D_MODEL * D_MODEL;   // 1,048,576

  bf16_t* xb  = (bf16_t*)alloc(nX * 2);
  bf16_t* wqb = (bf16_t*)alloc(nW * 2);
  bf16_t* wkb = (bf16_t*)alloc(nW * 2);
  bf16_t* wvb = (bf16_t*)alloc(nW * 2);
  bf16_t* wob = (bf16_t*)alloc(nW * 2);
  bf16_t* Qb  = (bf16_t*)alloc(nX * 2);          // (B,H,T,64)
  bf16_t* Kb  = (bf16_t*)alloc(nX * 2);          // (B,H,T,64)
  bf16_t* Vtb = (bf16_t*)alloc(nX * 2);          // (B,H,64,T)
  bf16_t* Ab  = (bf16_t*)alloc(nX * 2);          // (B,T,D) attention output
  // total ~48 MB of d_ws

  cvt_f32_bf16<<<(int)(nX / 1024), 256, 0, stream>>>(x,  xb,  (int)nX);
  cvt_f32_bf16<<<(int)(nW / 1024), 256, 0, stream>>>(Wq, wqb, (int)nW);
  cvt_f32_bf16<<<(int)(nW / 1024), 256, 0, stream>>>(Wk, wkb, (int)nW);
  cvt_f32_bf16<<<(int)(nW / 1024), 256, 0, stream>>>(Wv, wvb, (int)nW);
  cvt_f32_bf16<<<(int)(nW / 1024), 256, 0, stream>>>(Wo, wob, (int)nW);

  qkv_gemm<<<dim3(4, 64, 3), 256, 0, stream>>>(xb, wqb, wkb, wvb, Qb, Kb, Vtb);
  attn_kernel<<<dim3(32, 16, 2), 128, 0, stream>>>(Qb, Kb, Vtb, Ab);
  proj_gemm<<<dim3(4, 64), 256, 0, stream>>>(Ab, wob, (float*)d_out);
}
